// DCRNNModel_60696477827150
// MI455X (gfx1250) — matrix operand order repaired and checked by reading, not compiled, not run on hardware
//
#include <hip/hip_runtime.h>
#include <stdint.h>

#define NN 1024   // nodes
#define BB 64     // batch
#define UU 64     // units
#define QQ 5      // S*K+1
#define TT 12     // encoder steps
#define HH 12     // decoder steps

typedef __attribute__((ext_vector_type(16))) __bf16 bf16x16;
typedef __attribute__((ext_vector_type(8)))  float   f32x8;
typedef unsigned short bfraw;

union Frag {
  bfraw   h[16];
  uint4   q[2];
  bf16x16 v;
};

__device__ __forceinline__ bfraw f2bf(float f) {
  unsigned int u = __builtin_bit_cast(unsigned int, f);
  return (bfraw)((u + 0x7FFFu + ((u >> 16) & 1u)) >> 16);
}

// ---------------------------------------------------------------------------
// Diffusion GEMM:  C[(b,f), n] = alpha * sum_m A[(b,f), m] * sup[n, m]  (+ beta*X0)
// A: bf16 row-major (contiguous K). BT: support bf16 row-major [n][m], which is
// exactly the column-major storage the WMMA B-fragment wants (contiguous K).
// Wave computes a 32x64 tile (8 accumulators, 8 WMMA per k-step); fragment sets
// are double-buffered in registers so loads of step s+1 overlap WMMAs of step s.
// Block = 4 waves (2x2) -> 64x128 tile.  K must be a multiple of 64 (K = 1024).
// ---------------------------------------------------------------------------
__global__ __launch_bounds__(128) void gemm_diff(
    const bfraw* __restrict__ A, int lda,
    const bfraw* __restrict__ BT, int ldbt,
    float* __restrict__ C, bfraw* __restrict__ Cbf, int ldc,
    const float* __restrict__ X0, float alpha, float beta, int K)
{
  const int lane = threadIdx.x;
  const int wm = threadIdx.y & 1;
  const int wn = threadIdx.y >> 1;
  const int m0 = blockIdx.x * 64 + wm * 32;
  const int n0 = blockIdx.y * 128 + wn * 64;
  const int kh = lane >> 4;     // half-wave selects K sub-block per ISA layout
  const int lr = lane & 15;

  const bfraw* apr[2];
#pragma unroll
  for (int mi = 0; mi < 2; ++mi)
    apr[mi] = A + (long)(m0 + mi * 16 + lr) * lda + kh * 8;
  const bfraw* bpr[4];
#pragma unroll
  for (int j = 0; j < 4; ++j)
    bpr[j] = BT + (long)(n0 + j * 16 + lr) * ldbt + kh * 8;

  struct FragSet { Frag a[2]; Frag b[4]; };
  FragSet F0, F1;
  f32x8 acc[2][4] = {};

  auto loadFS = [&](FragSet& F, int k) {
#pragma unroll
    for (int mi = 0; mi < 2; ++mi) {
      F.a[mi].q[0] = *(const uint4*)(apr[mi] + k);       // K = k+kh*8 .. +7
      F.a[mi].q[1] = *(const uint4*)(apr[mi] + k + 16);  // K = k+16+kh*8 .. +7
    }
#pragma unroll
    for (int j = 0; j < 4; ++j) {
      F.b[j].q[0] = *(const uint4*)(bpr[j] + k);         // K = k+kh*8 .. +7
      F.b[j].q[1] = *(const uint4*)(bpr[j] + k + 16);    // K = k+16+kh*8 .. +7
    }
  };
  auto mma = [&](FragSet& F) {
#pragma unroll
    for (int mi = 0; mi < 2; ++mi)
#pragma unroll
      for (int j = 0; j < 4; ++j)
        acc[mi][j] = __builtin_amdgcn_wmma_f32_16x16x32_bf16(
            false, F.a[mi].v, false, F.b[j].v, (short)0, acc[mi][j], false, false);
  };

  loadFS(F0, 0);
  for (int k = 0; k < K; k += 64) {
    loadFS(F1, k + 32);
    mma(F0);
    if (k + 64 < K) loadFS(F0, k + 64);
    mma(F1);
  }

#pragma unroll
  for (int mi = 0; mi < 2; ++mi) {
    const int crow = m0 + mi * 16 + kh * 8;
#pragma unroll
    for (int j = 0; j < 4; ++j) {
      const int col = n0 + j * 16 + lr;
#pragma unroll
      for (int v = 0; v < 8; ++v) {
        long idx = (long)(crow + v) * ldc + col;
        float val = acc[mi][j][v];
        if (X0)  val = alpha * val + beta * X0[idx];   // Chebyshev: 2*A@x1 - x0
        if (C)   C[idx] = val;
        if (Cbf) Cbf[idx] = f2bf(val);
      }
    }
  }
}

// ---------------------------------------------------------------------------
// Cell GEMM (Q-block accumulate + fused activation epilogue):
//   C[b][o][n] = act( sum_q sum_j WT[q][o][j] * mats[q][b][j][n] + bias[o] )
// act: 0 = none, 1 = sigmoid, 2 = tanh.  WT contiguous along K; mats B-fragment
// uses strided 16-bit gathers (~10% of total flops). (q,k) space is flattened
// and double-buffered.
// ---------------------------------------------------------------------------
__global__ __launch_bounds__(128) void gemm_cell(
    const bfraw* __restrict__ WT,   // [Q][O][fpad]
    const bfraw* __restrict__ mats, // [Q][B][fpad][NN]
    float* __restrict__ C,          // [B][O][NN]
    const float* __restrict__ bias, int act,
    int O, int fpad)
{
  const int lane = threadIdx.x;
  const int b  = blockIdx.z;
  const int m0 = blockIdx.x * 16;
  const int n0 = blockIdx.y * 256 + threadIdx.y * 64;
  const int kh = lane >> 4;
  const int lr = lane & 15;

  struct FragSet { Frag a; Frag b[4]; };
  FragSet F0, F1;
  f32x8 acc[4] = {};

  const int kpq = fpad >> 5;         // 32-wide k-steps per q block
  const int nsteps = QQ * kpq;

  auto loadFS = [&](FragSet& F, int s) {
    int qd = s / kpq;
    int k  = (s - qd * kpq) << 5;
    const bfraw* wp = WT + ((long)qd * O + (m0 + lr)) * fpad + kh * 16 + k;
    F.a.q[0] = *(const uint4*)(wp);
    F.a.q[1] = *(const uint4*)(wp + 8);
    const bfraw* mp = mats + ((long)(qd * BB + b) * fpad + k + kh * 16) * NN;
#pragma unroll
    for (int j = 0; j < 4; ++j) {
      const bfraw* bp = mp + (n0 + j * 16 + lr);
#pragma unroll
      for (int i = 0; i < 16; ++i) F.b[j].h[i] = bp[(long)i * NN];
    }
  };
  auto mma = [&](FragSet& F) {
#pragma unroll
    for (int j = 0; j < 4; ++j)
      acc[j] = __builtin_amdgcn_wmma_f32_16x16x32_bf16(
          false, F.a.v, false, F.b[j].v, (short)0, acc[j], false, false);
  };

  loadFS(F0, 0);
  for (int s = 0; s < nsteps; s += 2) {
    if (s + 1 < nsteps) loadFS(F1, s + 1);
    mma(F0);
    if (s + 2 < nsteps) loadFS(F0, s + 2);
    if (s + 1 < nsteps) mma(F1);
  }

  const int crow = m0 + kh * 8;
#pragma unroll
  for (int j = 0; j < 4; ++j) {
    const int col = n0 + j * 16 + lr;
#pragma unroll
    for (int v = 0; v < 8; ++v) {
      float val = acc[j][v] + bias[crow + v];
      if (act == 1)      val = 1.f / (1.f + __expf(-val));
      else if (act == 2) val = tanhf(val);
      C[((long)b * O + crow + v) * NN + col] = val;
    }
  }
}

// ------------------------- elementwise helpers -----------------------------
__global__ void k_f32_to_bf16(const float* __restrict__ s, bfraw* __restrict__ d, long n) {
  long i = (long)blockIdx.x * blockDim.x + threadIdx.x;
  if (i < n) d[i] = f2bf(s[i]);
}

__global__ void k_zero_f32(float* __restrict__ p, long n) {
  long i = (long)blockIdx.x * blockDim.x + threadIdx.x;
  if (i < n) p[i] = 0.f;
}

// WT[q][o][j] = W[(q*f + j), o]  (zero pad rows j >= f), converted to bf16
__global__ void k_build_wt(const float* __restrict__ W, bfraw* __restrict__ WT,
                           int f, int fpad, int O) {
  long idx = (long)blockIdx.x * blockDim.x + threadIdx.x;
  long tot = (long)QQ * O * fpad;
  if (idx >= tot) return;
  int j = (int)(idx % fpad);
  long r = idx / fpad;
  int o = (int)(r % O);
  int q = (int)(r / O);
  float v = (j < f) ? W[((long)(q * f + j)) * O + o] : 0.f;
  WT[idx] = f2bf(v);
}

// Build concat input xcat[b][j][n] = [x | (r*)h | 0-pad] in f32 + bf16 (mats slot 0)
__global__ void k_concat(const float* __restrict__ inp, int t,
                         const float* __restrict__ xA, int xmode,
                         const float* __restrict__ hst,
                         const float* __restrict__ ru, int useRH,
                         float* __restrict__ xc, bfraw* __restrict__ m0,
                         int ind, int fpad)
{
  long idx = (long)blockIdx.x * blockDim.x + threadIdx.x;
  long tot = (long)BB * fpad * NN;
  if (idx >= tot) return;
  int n = (int)(idx & (NN - 1));
  long r = idx >> 10;
  int j = (int)(r % fpad);
  int b = (int)(r / fpad);
  float v = 0.f;
  if (j < ind) {
    if (xmode == 0)      v = inp[((long)b * NN + n) * TT + t];          // inputs[b,n,t]
    else if (xmode == 2) v = xA[(long)b * NN + n];                      // go[b,n]
    else                 v = xA[((long)b * UU + j) * NN + n];           // prev-layer h
  } else if (j < ind + UU) {
    int jj = j - ind;
    float hv = hst[((long)b * UU + jj) * NN + n];
    if (useRH) hv *= ru[((long)b * 128 + jj) * NN + n];                 // r * h
    v = hv;
  }
  xc[idx] = v;
  m0[idx] = f2bf(v);
}

// h = u*h + (1-u)*c   (c already tanh'd by gemm_cell epilogue; u sigmoid'd)
__global__ void k_hnew(float* __restrict__ h, const float* __restrict__ c,
                       const float* __restrict__ ru) {
  long idx = (long)blockIdx.x * blockDim.x + threadIdx.x;
  if (idx >= (long)BB * UU * NN) return;
  int n = (int)(idx & (NN - 1));
  int o = (int)((idx >> 10) & (UU - 1));
  int b = (int)(idx >> 16);
  float ug = ru[((long)b * 128 + UU + o) * NN + n];
  h[idx] = ug * h[idx] + (1.f - ug) * c[idx];
}

__global__ void k_proj(const float* __restrict__ h1, const float* __restrict__ pw,
                       const float* __restrict__ pb, float* __restrict__ o,
                       float* __restrict__ go) {
  long idx = (long)blockIdx.x * blockDim.x + threadIdx.x;
  if (idx >= (long)BB * NN) return;
  int n = (int)(idx & (NN - 1));
  int b = (int)(idx >> 10);
  float acc = pb[0];
#pragma unroll
  for (int j = 0; j < UU; ++j)
    acc += h1[((long)b * UU + j) * NN + n] * pw[j];
  o[idx]  = acc;
  go[idx] = acc;
}

// ---------------------------------------------------------------------------
extern "C" void kernel_launch(void* const* d_in, const int* in_sizes, int n_in,
                              void* d_out, int out_size, void* d_ws, size_t ws_size,
                              hipStream_t stream) {
  const float* inputs   = (const float*)d_in[0];
  const float* supports = (const float*)d_in[1];
  const float* enc_gw0 = (const float*)d_in[2];
  const float* enc_gb0 = (const float*)d_in[3];
  const float* enc_cw0 = (const float*)d_in[4];
  const float* enc_cb0 = (const float*)d_in[5];
  const float* enc_gw1 = (const float*)d_in[6];
  const float* enc_gb1 = (const float*)d_in[7];
  const float* enc_cw1 = (const float*)d_in[8];
  const float* enc_cb1 = (const float*)d_in[9];
  const float* dec_gw0 = (const float*)d_in[10];
  const float* dec_gb0 = (const float*)d_in[11];
  const float* dec_cw0 = (const float*)d_in[12];
  const float* dec_cb0 = (const float*)d_in[13];
  const float* dec_gw1 = (const float*)d_in[14];
  const float* dec_gb1 = (const float*)d_in[15];
  const float* dec_cw1 = (const float*)d_in[16];
  const float* dec_cb1 = (const float*)d_in[17];
  const float* proj_w  = (const float*)d_in[18];
  const float* proj_b  = (const float*)d_in[19];
  float* out = (float*)d_out;

  // ---- workspace bump allocator (~200 MB total) ----
  char* base = (char*)d_ws;
  size_t off = 0;
  auto alloc = [&](size_t bytes) -> void* {
    void* p = base + off;
    off += (bytes + 255) & ~(size_t)255;
    return p;
  };

  const int FP0 = 96;   // padded channels for f = 65 (layer 0)
  const int FP1 = 128;  // layer 1: f = 128
  bfraw* supbf = (bfraw*)alloc((size_t)2 * NN * NN * sizeof(bfraw));
  bfraw* mats  = (bfraw*)alloc((size_t)QQ * BB * FP1 * NN * sizeof(bfraw));
  float* xcat  = (float*)alloc((size_t)BB * FP1 * NN * sizeof(float));
  float* ruf   = (float*)alloc((size_t)BB * 128 * NN * sizeof(float));
  float* cpre  = (float*)alloc((size_t)BB * UU * NN * sizeof(float));
  float* h0    = (float*)alloc((size_t)BB * UU * NN * sizeof(float));
  float* h1    = (float*)alloc((size_t)BB * UU * NN * sizeof(float));
  float* go    = (float*)alloc((size_t)BB * NN * sizeof(float));
  bfraw* wt[8];
  const int wtO[8]  = {128, 64, 128, 64, 128, 64, 128, 64};
  const int wtF[8]  = {65, 65, 128, 128, 65, 65, 128, 128};
  const int wtFP[8] = {FP0, FP0, FP1, FP1, FP0, FP0, FP1, FP1};
  const float* wsrc[8] = {enc_gw0, enc_cw0, enc_gw1, enc_cw1,
                          dec_gw0, dec_cw0, dec_gw1, dec_cw1};
  for (int i = 0; i < 8; ++i)
    wt[i] = (bfraw*)alloc((size_t)QQ * wtO[i] * wtFP[i] * sizeof(bfraw));

  auto cdiv = [](long n, long d) { return (unsigned)((n + d - 1) / d); };

  // ---- one-time-per-launch prep (deterministic) ----
  k_f32_to_bf16<<<cdiv((long)2 * NN * NN, 256), 256, 0, stream>>>(
      supports, supbf, (long)2 * NN * NN);
  for (int i = 0; i < 8; ++i) {
    long tot = (long)QQ * wtO[i] * wtFP[i];
    k_build_wt<<<cdiv(tot, 256), 256, 0, stream>>>(wsrc[i], wt[i], wtF[i], wtFP[i], wtO[i]);
  }
  k_zero_f32<<<cdiv((long)BB * UU * NN, 256), 256, 0, stream>>>(h0, (long)BB * UU * NN);
  k_zero_f32<<<cdiv((long)BB * UU * NN, 256), 256, 0, stream>>>(h1, (long)BB * UU * NN);

  auto matq = [&](int q, int fpad) { return mats + (size_t)q * BB * fpad * NN; };

  auto diffusion = [&](int fpad) {
    long M = (long)BB * fpad;
    dim3 g((unsigned)(M / 64), NN / 128), blk(32, 4);
    for (int s = 0; s < 2; ++s) {
      const bfraw* sup = supbf + (size_t)s * NN * NN;
      // x1 = A_s @ x0   (bf16 result into mats slot 1+2s)
      gemm_diff<<<g, blk, 0, stream>>>(matq(0, fpad), NN, sup, NN,
                                       nullptr, matq(1 + 2 * s, fpad), NN,
                                       nullptr, 1.f, 0.f, NN);
      // x2 = 2*A_s @ x1 - x0  (bf16 result into mats slot 2+2s)
      gemm_diff<<<g, blk, 0, stream>>>(matq(1 + 2 * s, fpad), NN, sup, NN,
                                       nullptr, matq(2 + 2 * s, fpad), NN,
                                       xcat, 2.f, -1.f, NN);
    }
  };

  auto cell = [&](int xmode, int t, const float* xA, float* hst,
                  const bfraw* wtg, const float* gb,
                  const bfraw* wtc, const float* cb,
                  int ind, int fpad) {
    long cat = (long)BB * fpad * NN;
    // ru = sigmoid(diffusion([x,h]) @ gw + gb)  (activation fused in gemm_cell)
    k_concat<<<cdiv(cat, 256), 256, 0, stream>>>(inputs, t, xA, xmode, hst, ruf, 0,
                                                 xcat, matq(0, fpad), ind, fpad);
    diffusion(fpad);
    gemm_cell<<<dim3(128 / 16, NN / 256, BB), dim3(32, 4), 0, stream>>>(
        wtg, mats, ruf, gb, 1, 128, fpad);
    // c = tanh(diffusion([x, r*h]) @ cw + cb);  h = u*h + (1-u)*c
    k_concat<<<cdiv(cat, 256), 256, 0, stream>>>(inputs, t, xA, xmode, hst, ruf, 1,
                                                 xcat, matq(0, fpad), ind, fpad);
    diffusion(fpad);
    gemm_cell<<<dim3(UU / 16, NN / 256, BB), dim3(32, 4), 0, stream>>>(
        wtc, mats, cpre, cb, 2, UU, fpad);
    k_hnew<<<cdiv((long)BB * UU * NN, 256), 256, 0, stream>>>(hst, cpre, ruf);
  };

  // ---- encoder ----
  for (int t = 0; t < TT; ++t) {
    cell(0, t, nullptr, h0, wt[0], enc_gb0, wt[1], enc_cb0, 1, FP0);
    cell(1, 0, h0,      h1, wt[2], enc_gb1, wt[3], enc_cb1, UU, FP1);
  }
  // ---- decoder ----
  k_zero_f32<<<cdiv((long)BB * NN, 256), 256, 0, stream>>>(go, (long)BB * NN);
  for (int s = 0; s < HH; ++s) {
    cell(2, 0, go, h0, wt[4], dec_gb0, wt[5], dec_cb0, 1, FP0);
    cell(1, 0, h0, h1, wt[6], dec_gb1, wt[7], dec_cb1, UU, FP1);
    k_proj<<<cdiv((long)BB * NN, 256), 256, 0, stream>>>(
        h1, proj_w, proj_b, out + (size_t)s * BB * NN, go);
  }

  (void)in_sizes; (void)n_in; (void)out_size; (void)ws_size;
}
